// ConditionalRandomFieldModel_52278341927591
// MI455X (gfx1250) — matrix operand order repaired and checked by reading, not compiled, not run on hardware
//
#include <hip/hip_runtime.h>
#include <hip/hip_bf16.h>

#define D_ 256
#define K_TAGS 50
#define NEGV -1000000000.0f

typedef __attribute__((ext_vector_type(16))) __bf16 v16bf;
typedef __attribute__((ext_vector_type(8)))  float  v8f;
typedef unsigned int u32x4 __attribute__((ext_vector_type(4)));
typedef float        f32x4 __attribute__((ext_vector_type(4)));

union ABfrag { v16bf v; u32x4 q[2]; unsigned int u[8]; };

__device__ __forceinline__ unsigned short f2bf(float f) {
    unsigned int u = __float_as_uint(f);
    unsigned int r = u + 0x7FFFu + ((u >> 16) & 1u);
    return (unsigned short)(r >> 16);
}
__device__ __forceinline__ float sigmoidf_(float x) {
    return 1.f / (1.f + __expf(-x));
}

// ---------------- prep kernels ----------------

// Pack weight W (N_out rows x ldw cols, submatrix at col0, K cols) into WMMA B-fragment
// register order: dword id = ((kblk*Ntiles + tn)*32 + lane)*8 + v
// holds bf16{ B[kb+2v, n], B[kb+2v+1, n] } with kb = kblk*32 + (lane>>4)*16, n = tn*16 + (lane&15),
// where B[k,n] = W[n, col0+k].
__global__ void packB_kernel(const float* __restrict__ W, unsigned int* __restrict__ Bp,
                             int K, int Ntiles, int ldw, int col0) {
    int id = blockIdx.x * 256 + threadIdx.x;
    int total = (K >> 5) * Ntiles * 256;
    if (id >= total) return;
    int v    = id & 7;
    int lane = (id >> 3) & 31;
    int tn   = (id >> 8) % Ntiles;
    int kblk = (id >> 8) / Ntiles;
    int kb = kblk * 32 + (lane >> 4) * 16 + 2 * v;
    int nn = tn * 16 + (lane & 15);
    unsigned int lo = f2bf(W[(size_t)nn * ldw + col0 + kb]);
    unsigned int hi = f2bf(W[(size_t)nn * ldw + col0 + kb + 1]);
    Bp[id] = lo | (hi << 16);
}

// R[s*cols + d] = W[d*ldw + col0 + s]
__global__ void rowext_kernel(const float* __restrict__ W, float* __restrict__ R,
                              int rows, int cols, int ldw, int col0) {
    int idx = blockIdx.x * 256 + threadIdx.x;
    if (idx >= rows * cols) return;
    int s = idx / cols, d = idx % cols;
    R[idx] = W[(size_t)d * ldw + col0 + s];
}

__global__ void bias2_kernel(const float* a, const float* b, float* o) {
    int i = threadIdx.x; o[i] = a[i] + b[i];
}

// xpad (Mpx x 256 bf16): rows 2..n+1 = E[words], else 0.  Ew (Mpw x 256): rows 0..n-1 = E[words]
__global__ void embed_kernel(const float* __restrict__ E, const int* __restrict__ words,
                             unsigned short* __restrict__ xpad, unsigned short* __restrict__ Ew,
                             int n, int Mpx, int Mpw) {
    int row = blockIdx.x, col = threadIdx.x;
    if (row < Mpx) {
        float v = 0.f;
        if (row >= 2 && row < n + 2) v = E[(size_t)words[row - 2] * D_ + col];
        xpad[(size_t)row * D_ + col] = f2bf(v);
    }
    if (row < Mpw) {
        float v = (row < n) ? E[(size_t)words[row] * D_ + col] : 0.f;
        Ew[(size_t)row * D_ + col] = f2bf(v);
    }
}

// ---------------- WMMA GEMM:  C[M,N] = A[M,KK](bf16) * B[KK,N](pre-packed) + bias[n] ----------
// wave computes one 16x16 tile. A: 2x b128 per K-step; B: 2x b128 from fragment-packed buffer.
template<int KK>
__global__ void gemm_bf16_wmma(const unsigned short* __restrict__ A,
                               const unsigned int* __restrict__ Bp,
                               const float* __restrict__ bias,
                               float* __restrict__ C, int N) {
    int wave = threadIdx.x >> 5;
    int lane = threadIdx.x & 31;
    int tm = blockIdx.x;
    int Ntiles = N >> 4;
    int tn = blockIdx.y * 8 + wave;
    if (tn >= Ntiles) return;
    int half = lane >> 4;
    int l16  = lane & 15;
    const unsigned short* Arow = A + (size_t)(tm * 16 + l16) * KK;
    const u32x4* Bq = (const u32x4*)Bp + ((size_t)tn * 32 + lane) * 2;
    const size_t bstride = (size_t)Ntiles * 64;   // u32x4 per 32-K block
    int nb = tn * 16 + l16;
    v8f acc = {0.f, 0.f, 0.f, 0.f, 0.f, 0.f, 0.f, 0.f};
    #pragma unroll 4
    for (int k0 = 0; k0 < KK; k0 += 32) {
        __builtin_prefetch(Arow + k0 + 128, 0, 1);   // global_prefetch_b8
        ABfrag a, b;
        a.q[0] = *(const u32x4*)(Arow + k0 + half * 16);
        a.q[1] = *(const u32x4*)(Arow + k0 + half * 16 + 8);
        b.q[0] = Bq[0];
        b.q[1] = Bq[1];
        Bq += bstride;
        acc = __builtin_amdgcn_wmma_f32_16x16x32_bf16(
            false, a.v, false, b.v, (short)0, acc, false, false);
    }
    float bn = bias ? bias[nb] : 0.f;
    #pragma unroll
    for (int v = 0; v < 8; ++v) {
        int m = tm * 16 + half * 8 + v;
        C[(size_t)m * N + nb] = acc[v] + bn;
    }
}

// ---------------- persistent RNN scan, one block per direction ----------------
__global__ void rnn_kernel(const float* __restrict__ pre_f, const float* __restrict__ Whh_f,
                           float* __restrict__ hf,
                           const float* __restrict__ pre_b, const float* __restrict__ Whh_b,
                           float* __restrict__ hb, int T) {
    int rev = blockIdx.x;
    const float* pre = rev ? pre_b : pre_f;
    const float* Whh = rev ? Whh_b : Whh_f;
    float* out       = rev ? hb : hf;
    __shared__ __align__(16) float h[D_];
    int o = threadIdx.x;
    h[o] = 0.f;
    __syncthreads();
    const f32x4* wr4 = (const f32x4*)(Whh + (size_t)o * D_);
    const f32x4* h4  = (const f32x4*)h;
    for (int step = 0; step < T; ++step) {
        int t = rev ? (T - 1 - step) : step;
        float acc = pre[(size_t)t * D_ + o];
        #pragma unroll 8
        for (int i = 0; i < D_ / 4; ++i) {
            f32x4 w = wr4[i];
            f32x4 hh = h4[i];
            acc += w[0] * hh[0] + w[1] * hh[1] + w[2] * hh[2] + w[3] * hh[3];
        }
        float nh = tanhf(acc);
        __syncthreads();
        h[o] = nh;
        out[(size_t)t * D_ + o] = nh;
        __syncthreads();
    }
}

// ---------------- sigmoid(h), build a_h and hcat (bf16, padded) ----------------
__global__ void post_kernel(const float* __restrict__ hf, const float* __restrict__ hb,
                            unsigned short* __restrict__ a_h, unsigned short* __restrict__ hcat,
                            int n) {
    int row = blockIdx.x;
    int col = blockIdx.y * 256 + threadIdx.x;   // 0..511
    float va = 0.f, vc = 0.f;
    if (row < n + 2) {
        if (col < D_) {
            va = sigmoidf_(hf[(size_t)row * D_ + col]);
            vc = sigmoidf_(hf[(size_t)(row + 1) * D_ + col]);
        } else {
            int c = col - D_;
            va = sigmoidf_(hb[(size_t)(row + 1) * D_ + c]);
            vc = sigmoidf_(hb[(size_t)(row + 1) * D_ + c]);
        }
    }
    a_h [(size_t)row * 512 + col] = f2bf(va);
    hcat[(size_t)row * 512 + col] = f2bf(vc);
}

// ---------------- A tensor: (n+1) x 50 x 50, 256-length sigmoid reduction ----------------
__global__ void a_kernel(const float* __restrict__ base_a, const float* __restrict__ WsT,
                         const float* __restrict__ WtT, const float* __restrict__ theta,
                         const int* __restrict__ bosp, float* __restrict__ Abuf, int n) {
    __shared__ __align__(16) float sA[D_], sTh[D_];
    int j = blockIdx.x + 1;                 // j = 1..n+1
    sA [threadIdx.x] = base_a[(size_t)j * D_ + threadIdx.x];
    sTh[threadIdx.x] = theta[threadIdx.x];
    __syncthreads();
    int idx = blockIdx.y * 256 + threadIdx.x;
    if (idx >= K_TAGS * K_TAGS) return;
    int s = idx / K_TAGS, t = idx % K_TAGS;
    const f32x4* ws4 = (const f32x4*)(WsT + s * D_);
    const f32x4* wt4 = (const f32x4*)(WtT + t * D_);
    const f32x4* sA4 = (const f32x4*)sA;
    const f32x4* th4 = (const f32x4*)sTh;
    float acc = 0.f;
    #pragma unroll 2
    for (int i = 0; i < D_ / 4; ++i) {
        f32x4 x = sA4[i] + ws4[i] + wt4[i];
        f32x4 th = th4[i];
        acc += th[0] * sigmoidf_(x[0]) + th[1] * sigmoidf_(x[1]) +
               th[2] * sigmoidf_(x[2]) + th[3] * sigmoidf_(x[3]);
    }
    int bos = *bosp;
    Abuf[(size_t)(j - 1) * (K_TAGS * K_TAGS) + idx] = (t == bos) ? 0.f : acc;
}

// ---------------- Bo: n x 50 ----------------
__global__ void b_kernel(const float* __restrict__ base_b, const float* __restrict__ WtbT,
                         const float* __restrict__ wcon, const float* __restrict__ theta,
                         const int* __restrict__ bosp, const int* __restrict__ eosp,
                         float* __restrict__ Bo, int n) {
    int j = blockIdx.x;                     // 0..n-1
    __shared__ __align__(16) float sB[D_], sW[D_], sTh[D_];
    sB [threadIdx.x] = base_b[(size_t)(j + 1) * D_ + threadIdx.x];
    sW [threadIdx.x] = wcon[(size_t)j * D_ + threadIdx.x];
    sTh[threadIdx.x] = theta[threadIdx.x];
    __syncthreads();
    int t = threadIdx.x;
    if (t < K_TAGS) {
        const f32x4* wt4 = (const f32x4*)(WtbT + t * D_);
        const f32x4* sB4 = (const f32x4*)sB;
        const f32x4* sW4 = (const f32x4*)sW;
        const f32x4* th4 = (const f32x4*)sTh;
        float acc = 0.f;
        #pragma unroll 2
        for (int i = 0; i < D_ / 4; ++i) {
            f32x4 x = sB4[i] + wt4[i] + sW4[i];
            f32x4 th = th4[i];
            acc += th[0] * sigmoidf_(x[0]) + th[1] * sigmoidf_(x[1]) +
                   th[2] * sigmoidf_(x[2]) + th[3] * sigmoidf_(x[3]);
        }
        int bos = *bosp, eos = *eosp;
        Bo[(size_t)j * K_TAGS + t] = (t == bos || t == eos) ? 0.f : acc;
    }
}

// ---------------- CRF forward scan (single block) ----------------
__global__ void crf_kernel(const float* __restrict__ Abuf, const float* __restrict__ Bo,
                           int n, const int* __restrict__ bosp, const int* __restrict__ eosp,
                           float* __restrict__ out) {
    __shared__ float alpha[64];
    int t = threadIdx.x;
    int bos = *bosp, eos = *eosp;
    alpha[t] = (t == bos) ? 0.f : NEGV;
    __syncthreads();
    for (int j = 1; j <= n; ++j) {
        const float* Aj = Abuf + (size_t)(j - 1) * (K_TAGS * K_TAGS);
        float nv = 0.f;
        if (t < K_TAGS) {
            float m = -3.4e38f;
            for (int s = 0; s < K_TAGS; ++s) m = fmaxf(m, alpha[s] + Aj[s * K_TAGS + t]);
            float sum = 0.f;
            for (int s = 0; s < K_TAGS; ++s) sum += __expf(alpha[s] + Aj[s * K_TAGS + t] - m);
            nv = Bo[(size_t)(j - 1) * K_TAGS + t] + m + __logf(sum);
        }
        __syncthreads();
        if (t < K_TAGS) alpha[t] = nv;
        __syncthreads();
    }
    if (t == 0) {
        const float* Aj = Abuf + (size_t)n * (K_TAGS * K_TAGS);   // j = n+1
        float m = -3.4e38f;
        for (int s = 0; s < K_TAGS; ++s) m = fmaxf(m, alpha[s] + Aj[s * K_TAGS + eos]);
        float sum = 0.f;
        for (int s = 0; s < K_TAGS; ++s) sum += __expf(alpha[s] + Aj[s * K_TAGS + eos] - m);
        out[0] = m + __logf(sum);
    }
}

// ---------------- host launch ----------------
static inline size_t ru16(size_t v) { return (v + 15) & ~(size_t)15; }

extern "C" void kernel_launch(void* const* d_in, const int* in_sizes, int n_in,
                              void* d_out, int out_size, void* d_ws, size_t ws_size,
                              hipStream_t stream) {
    const float* E     = (const float*)d_in[0];
    const float* Wih_f = (const float*)d_in[1];
    const float* Whh_f = (const float*)d_in[2];
    const float* bih_f = (const float*)d_in[3];
    const float* bhh_f = (const float*)d_in[4];
    const float* Wih_b = (const float*)d_in[5];
    const float* Whh_b = (const float*)d_in[6];
    const float* bih_b = (const float*)d_in[7];
    const float* bhh_b = (const float*)d_in[8];
    const float* Wa    = (const float*)d_in[9];
    const float* ba    = (const float*)d_in[10];
    const float* Wb    = (const float*)d_in[11];
    const float* bb    = (const float*)d_in[12];
    const float* ThA   = (const float*)d_in[13];
    const float* ThB   = (const float*)d_in[14];
    const int*   words = (const int*)d_in[15];
    const int*   bosp  = (const int*)d_in[16];
    const int*   eosp  = (const int*)d_in[17];
    float* out = (float*)d_out;

    const int n    = in_sizes[15];
    const int ldwA = in_sizes[9]  / D_;   // 612
    const int ldwB = in_sizes[11] / D_;   // 818
    const int Mpx = (int)ru16(n + 3);
    const int Mpa = (int)ru16(n + 2);
    const int Mpw = (int)ru16(n);
    const int Ntiles = D_ / 16;           // 16

    // workspace carve-up
    char* base = (char*)d_ws;
    size_t off = 0;
    auto alloc = [&](size_t bytes) -> char* {
        char* p = base + off;
        off += (bytes + 255) & ~(size_t)255;
        return p;
    };
    unsigned short* xpad  = (unsigned short*)alloc((size_t)Mpx * D_ * 2);
    unsigned short* Ew    = (unsigned short*)alloc((size_t)Mpw * D_ * 2);
    unsigned int* BpWihf  = (unsigned int*)alloc((size_t)D_ * D_ * 2);   // K*N/2 dwords
    unsigned int* BpWihb  = (unsigned int*)alloc((size_t)D_ * D_ * 2);
    unsigned int* BpWa    = (unsigned int*)alloc((size_t)512 * D_ * 2);
    unsigned int* BpWb    = (unsigned int*)alloc((size_t)512 * D_ * 2);
    unsigned int* BpWww   = (unsigned int*)alloc((size_t)D_ * D_ * 2);
    float* WsT    = (float*)alloc((size_t)K_TAGS * D_ * 4);
    float* WtT    = (float*)alloc((size_t)K_TAGS * D_ * 4);
    float* WtbT   = (float*)alloc((size_t)K_TAGS * D_ * 4);
    float* bsum_f = (float*)alloc(D_ * 4);
    float* bsum_b = (float*)alloc(D_ * 4);
    float* pre_f  = (float*)alloc((size_t)Mpx * D_ * 4);
    float* pre_b  = (float*)alloc((size_t)Mpx * D_ * 4);
    float* hf     = (float*)alloc((size_t)Mpx * D_ * 4);
    float* hb     = (float*)alloc((size_t)Mpx * D_ * 4);
    unsigned short* a_h  = (unsigned short*)alloc((size_t)Mpa * 512 * 2);
    unsigned short* hcat = (unsigned short*)alloc((size_t)Mpa * 512 * 2);
    float* base_a = (float*)alloc((size_t)Mpa * D_ * 4);
    float* base_b = (float*)alloc((size_t)Mpa * D_ * 4);
    float* wcon   = (float*)alloc((size_t)Mpw * D_ * 4);
    float* Abuf   = (float*)alloc((size_t)(n + 1) * K_TAGS * K_TAGS * 4);
    float* Bo     = (float*)alloc((size_t)n * K_TAGS * 4);
    (void)ws_size;

    // --- prep: pack weights into WMMA B-fragment order, gather embeddings ---
    int g;
    g = ((256 >> 5) * Ntiles * 256 + 255) / 256;       // K=256 packs
    packB_kernel<<<g, 256, 0, stream>>>(Wih_f, BpWihf, 256, Ntiles, D_, 0);
    packB_kernel<<<g, 256, 0, stream>>>(Wih_b, BpWihb, 256, Ntiles, D_, 0);
    packB_kernel<<<g, 256, 0, stream>>>(Wb, BpWww, 256, Ntiles, ldwB, 2 * D_ + K_TAGS);
    g = ((512 >> 5) * Ntiles * 256 + 255) / 256;       // K=512 packs
    packB_kernel<<<g, 256, 0, stream>>>(Wa, BpWa, 512, Ntiles, ldwA, 0);
    packB_kernel<<<g, 256, 0, stream>>>(Wb, BpWb, 512, Ntiles, ldwB, 0);
    g = (K_TAGS * D_ + 255) / 256;
    rowext_kernel<<<g, 256, 0, stream>>>(Wa, WsT,  K_TAGS, D_, ldwA, 2 * D_);
    rowext_kernel<<<g, 256, 0, stream>>>(Wa, WtT,  K_TAGS, D_, ldwA, 2 * D_ + K_TAGS);
    rowext_kernel<<<g, 256, 0, stream>>>(Wb, WtbT, K_TAGS, D_, ldwB, 2 * D_);
    bias2_kernel<<<1, 256, 0, stream>>>(bih_f, bhh_f, bsum_f);
    bias2_kernel<<<1, 256, 0, stream>>>(bih_b, bhh_b, bsum_b);
    embed_kernel<<<(Mpx > Mpw ? Mpx : Mpw), 256, 0, stream>>>(E, words, xpad, Ew, n, Mpx, Mpw);

    // --- batched input GEMMs (WMMA bf16): pre = xpad @ Wih^T + (bih+bhh) ---
    dim3 blk(256);
    gemm_bf16_wmma<256><<<dim3(Mpx / 16, 2), blk, 0, stream>>>(xpad, BpWihf, bsum_f, pre_f, D_);
    gemm_bf16_wmma<256><<<dim3(Mpx / 16, 2), blk, 0, stream>>>(xpad, BpWihb, bsum_b, pre_b, D_);
    // wcon = E[words] @ Www^T  (independent of RNN)
    gemm_bf16_wmma<256><<<dim3(Mpw / 16, 2), blk, 0, stream>>>(Ew, BpWww, nullptr, wcon, D_);

    // --- sequential BiRNN (both directions concurrently) ---
    rnn_kernel<<<2, 256, 0, stream>>>(pre_f, Whh_f, hf, pre_b, Whh_b, hb, n + 3);

    // --- sigmoid + concat ---
    post_kernel<<<dim3(Mpa, 2), 256, 0, stream>>>(hf, hb, a_h, hcat, n);

    // --- base_a / base_b GEMMs (WMMA bf16, K=512) ---
    gemm_bf16_wmma<512><<<dim3(Mpa / 16, 2), blk, 0, stream>>>(a_h,  BpWa, ba, base_a, D_);
    gemm_bf16_wmma<512><<<dim3(Mpa / 16, 2), blk, 0, stream>>>(hcat, BpWb, bb, base_b, D_);

    // --- A tensor (dominant throughput term) and Bo ---
    a_kernel<<<dim3(n + 1, (K_TAGS * K_TAGS + 255) / 256), 256, 0, stream>>>(
        base_a, WsT, WtT, ThA, bosp, Abuf, n);
    b_kernel<<<n, 256, 0, stream>>>(base_b, WtbT, wcon, ThB, bosp, eosp, Bo, n);

    // --- CRF forward scan ---
    crf_kernel<<<1, 64, 0, stream>>>(Abuf, Bo, n, bosp, eosp, out);
}
